// ForwardDeformer_86577950753311
// MI455X (gfx1250) — matrix operand order repaired and checked by reading, not compiled, not run on hardware
//
#include <hip/hip_runtime.h>
#include <hip/hip_bf16.h>
#include <math.h>

typedef __attribute__((ext_vector_type(16))) _Float16 v16h;
typedef __attribute__((ext_vector_type(8)))  _Float16 v8h;
typedef __attribute__((ext_vector_type(8)))  float    v8f;
typedef __attribute__((ext_vector_type(4)))  float    v4f;

#define SOFT_BLEND   20.0f
#define MAX_STEPS    20
#define CVG_THRESH   1e-5f
#define DVG_THRESH   1.0f
#define BR_EPS       1e-6f
#define NBONES       24
#define NINIT        9
#define HID          64
#define NW           4                 // waves per block
#define CAND_PER_WAVE 16
#define CAND_PER_BLOCK (NW * CAND_PER_WAVE)

#define LOG2E  1.4426950408889634f
#define LN2    0.6931471805599453f
#define SB_LOG2E (SOFT_BLEND * LOG2E)   // softmax scale folded into exp2

__constant__ int c_init_bones[NINIT] = {0, 1, 2, 4, 5, 16, 17, 18, 19};

__device__ __forceinline__ void lds_fence_wave() {
    // LDS is in-order per wave; this guarantees prior ds stores are visible
    // to subsequent ds loads and blocks compiler reordering.
    asm volatile("s_wait_dscnt 0" ::: "memory");
}

__device__ __forceinline__ float fast_rcp(float x) {
    return __builtin_amdgcn_rcpf(x);   // v_rcp_f32
}
__device__ __forceinline__ float fast_exp2(float x) {
    return __builtin_amdgcn_exp2f(x);  // v_exp_f32
}
__device__ __forceinline__ float fast_log2(float x) {
    return __builtin_amdgcn_logf(x);   // v_log_f32 (arg in (1,2], no guards needed)
}

__device__ __forceinline__ v8f wmma16(v16h a, v16h b, v8f c) {
    // D = A(16x32 f16) * B(32x16 f16) + C(16x16 f32)
    return __builtin_amdgcn_wmma_f32_16x16x32_f16(
        /*neg_a=*/false, a, /*neg_b=*/false, b,
        /*c_mod=*/(short)0, c, /*reuse_a=*/false, /*reuse_b=*/false);
}

// A-matrix fragment (16x32 f16, M = lane&15), ISA interleaved K layout:
// lanes 0-15 hold K = ko+{0..7, 16..23}; lanes 16-31 hold K = ko+{8..15, 24..31}
__device__ __forceinline__ v16h frag_ld_a(const _Float16* base, int row, int ko, int lane) {
    const _Float16* r = base + row * HID + ko + ((lane < 16) ? 0 : 8);
    v8h h0 = *(const v8h*)(r);
    v8h h1 = *(const v8h*)(r + 16);
    v16h a;
#pragma unroll
    for (int j = 0; j < 8; ++j) { a[j] = h0[j]; a[8 + j] = h1[j]; }
    return a;
}

// B-matrix fragment (32x16 f16, N = lane&15), ISA interleaved K layout:
// lanes 0-15 hold K = ko+{0..7, 16..23}; lanes 16-31 hold K = ko+{8..15, 24..31}
// base points at a row-major [N][K] (transposed-weight) array with stride HID.
__device__ __forceinline__ v16h frag_ld_b(const _Float16* base, int row, int ko, int lane) {
    const _Float16* r = base + row * HID + ko + ((lane < 16) ? 0 : 8);
    v8h h0 = *(const v8h*)(r);
    v8h h1 = *(const v8h*)(r + 16);
    v16h b;
#pragma unroll
    for (int j = 0; j < 8; ++j) { b[j] = h0[j]; b[8 + j] = h1[j]; }
    return b;
}

// Raw-trans softplus + sigmoid: 1+e is in (1,2], so unguarded v_log_f32 is exact enough.
__device__ __forceinline__ void softplus_sig(float z, float& sp, float& sg) {
    float e  = fast_exp2(-LOG2E * fabsf(z));   // exp(-|z|)
    float pe = 1.f + e;
    sp = fmaf(LN2, fast_log2(pe), fmaxf(z, 0.f));
    float d = fast_rcp(pe);
    sg = (z >= 0.f) ? d : (1.f - d);
}

__global__ __launch_bounds__(NW * 32)
void fd_broyden_kernel(const float* __restrict__ xd, const float* __restrict__ tfs,
                       const float* __restrict__ W0, const float* __restrict__ b0,
                       const float* __restrict__ W1, const float* __restrict__ b1,
                       const float* __restrict__ W2, const float* __restrict__ b2,
                       float* __restrict__ out, int P) {
    // ---- shared memory ----
    __shared__ __align__(16) _Float16 sW1T[HID * HID];   // [n][k] = W1[k][n]
    __shared__ __align__(16) _Float16 sW2T[32 * HID];    // [n][k] = W2[k][n], n>=24 zero
    __shared__ float sW0[4 * HID];                       // rows 0..2 = W0, row 3 = b0
    __shared__ float sB1[HID];
    __shared__ float sB2[NBONES];
    __shared__ __align__(16) float sTf[NBONES * 12];     // upper 3x4 of each bone tf
    __shared__ __align__(16) float sTinv[NINIT * 12];    // affine inverse of init bones
    __shared__ __align__(16) float sX[NW][16 * 4];       // current x per candidate
    __shared__ __align__(16) _Float16 sH[NW][16 * HID];  // activation transpose buffer
    __shared__ __align__(16) _Float16 sS0[NW][16 * HID]; // sigmoid(z0)
    __shared__ __align__(16) _Float16 sS1[NW][16 * HID]; // sigmoid(z1)
    __shared__ __align__(16) float sG[NW][16 * NBONES];  // logits / tangent logits

    const int tid = threadIdx.x;

    // ---- one-time staging (f32 -> f16, transposes) ----
    for (int idx = tid; idx < HID * HID; idx += blockDim.x) {
        int n = idx >> 6, k = idx & 63;
        sW1T[idx] = (_Float16)W1[k * HID + n];
    }
    for (int idx = tid; idx < 32 * HID; idx += blockDim.x) {
        int n = idx >> 6, k = idx & 63;
        sW2T[idx] = (_Float16)((n < NBONES) ? W2[k * NBONES + n] : 0.f);
    }
    for (int idx = tid; idx < 4 * HID; idx += blockDim.x) {
        int k = idx >> 6, n = idx & 63;
        sW0[idx] = (k < 3) ? W0[k * HID + n] : b0[n];
    }
    if (tid < HID) sB1[tid] = b1[tid];
    if (tid < NBONES) sB2[tid] = b2[tid];
    for (int idx = tid; idx < NBONES * 12; idx += blockDim.x) {
        int j = idx / 12, e = idx % 12;
        sTf[idx] = tfs[j * 16 + e];
    }
    if (tid < NINIT) {
        const float* T = tfs + c_init_bones[tid] * 16;
        float a00 = T[0], a01 = T[1], a02 = T[2],  t0 = T[3];
        float a10 = T[4], a11 = T[5], a12 = T[6],  t1 = T[7];
        float a20 = T[8], a21 = T[9], a22 = T[10], t2 = T[11];
        float det = a00 * (a11 * a22 - a12 * a21)
                  - a01 * (a10 * a22 - a12 * a20)
                  + a02 * (a10 * a21 - a11 * a20);
        float id = fast_rcp(det);
        float i00 = (a11 * a22 - a12 * a21) * id, i01 = (a02 * a21 - a01 * a22) * id, i02 = (a01 * a12 - a02 * a11) * id;
        float i10 = (a12 * a20 - a10 * a22) * id, i11 = (a00 * a22 - a02 * a20) * id, i12 = (a02 * a10 - a00 * a12) * id;
        float i20 = (a10 * a21 - a11 * a20) * id, i21 = (a01 * a20 - a00 * a21) * id, i22 = (a00 * a11 - a01 * a10) * id;
        float* O = sTinv + tid * 12;
        O[0] = i00; O[1] = i01; O[2]  = i02; O[3]  = -(i00 * t0 + i01 * t1 + i02 * t2);
        O[4] = i10; O[5] = i11; O[6]  = i12; O[7]  = -(i10 * t0 + i11 * t1 + i12 * t2);
        O[8] = i20; O[9] = i21; O[10] = i22; O[11] = -(i20 * t0 + i21 * t1 + i22 * t2);
    }
    __syncthreads();

    const int lane = tid & 31;
    const int wv   = tid >> 5;
    const int m    = lane & 15;          // candidate slot / A-matrix row
    const int nl   = lane & 15;          // B/C/D column within tile
    const int mh   = (lane >> 4) << 3;   // C/D row base for this half-wave

    // ---- candidate init: xc = (T_inv[bone] @ [xd,1])[:3]  (state mirrored in lane m and 16+m) ----
    const int cand = blockIdx.x * CAND_PER_BLOCK + wv * CAND_PER_WAVE + m;
    const int cc   = (cand < P) ? cand : (P - 1);
    const int pn   = cc / NINIT;
    const int bi   = cc - pn * NINIT;

    float T0 = xd[pn * 3 + 0], T1 = xd[pn * 3 + 1], T2 = xd[pn * 3 + 2];
    const float* Ti = sTinv + bi * 12;
    float X0 = Ti[0] * T0 + Ti[1] * T1 + Ti[2]  * T2 + Ti[3];
    float X1 = Ti[4] * T0 + Ti[5] * T1 + Ti[6]  * T2 + Ti[7];
    float X2 = Ti[8] * T0 + Ti[9] * T1 + Ti[10] * T2 + Ti[11];

    float wgt[NBONES];   // softmax weights at x_init (for Jacobian)
    float Rb[9];         // blended rotation at x_init

    // ---- wave-cooperative MLP + skinning eval; logits end up in sG ----
    // B fragments are re-read from LDS per eval (keeps VGPR count down for occupancy).
    auto mlp_eval = [&](bool save) {
        v4f xr[8];
#pragma unroll
        for (int v = 0; v < 8; ++v)
            xr[v] = *(const v4f*)(&sX[wv][(v + mh) * 4]);   // one b128 per row
        // layer 0 (K=3 + bias) in VALU, produced directly in C-layout
#pragma unroll
        for (int nt = 0; nt < 4; ++nt) {
            int n = nl + 16 * nt;
            float w00 = sW0[n], w01 = sW0[64 + n], w02 = sW0[128 + n], w03 = sW0[192 + n];
#pragma unroll
            for (int v = 0; v < 8; ++v) {
                int mm = v + mh;
                float z = w03 + xr[v].x * w00 + xr[v].y * w01 + xr[v].z * w02;
                float sp, sg; softplus_sig(z, sp, sg);
                sH[wv][mm * HID + n] = (_Float16)sp;
                if (save) sS0[wv][mm * HID + n] = (_Float16)sg;
            }
        }
        lds_fence_wave();
        // layer 1: 16x64x64 via WMMA
        v16h a0 = frag_ld_a(sH[wv], m, 0, lane);
        v16h a1 = frag_ld_a(sH[wv], m, 32, lane);
#pragma unroll
        for (int nt = 0; nt < 4; ++nt) {
            v8f acc = {};
            acc = wmma16(a0, frag_ld_b(sW1T, nl + 16 * nt, 0, lane), acc);
            acc = wmma16(a1, frag_ld_b(sW1T, nl + 16 * nt, 32, lane), acc);
            int n = nl + 16 * nt;
            float bb = sB1[n];
#pragma unroll
            for (int v = 0; v < 8; ++v) {
                int mm = v + mh;
                float z = acc[v] + bb;
                float sp, sg; softplus_sig(z, sp, sg);
                sH[wv][mm * HID + n] = (_Float16)sp;
                if (save) sS1[wv][mm * HID + n] = (_Float16)sg;
            }
        }
        lds_fence_wave();
        // layer 2: 16x32x64 via WMMA (N padded 24 -> 32 with zero weights)
        v16h c0 = frag_ld_a(sH[wv], m, 0, lane);
        v16h c1 = frag_ld_a(sH[wv], m, 32, lane);
#pragma unroll
        for (int nt = 0; nt < 2; ++nt) {
            v8f acc = {};
            acc = wmma16(c0, frag_ld_b(sW2T, nl + 16 * nt, 0, lane), acc);
            acc = wmma16(c1, frag_ld_b(sW2T, nl + 16 * nt, 32, lane), acc);
            int n = nl + 16 * nt;
            if (n < NBONES) {
                float bb = sB2[n];
#pragma unroll
                for (int v = 0; v < 8; ++v)
                    sG[wv][(v + mh) * NBONES + n] = acc[v] + bb;
            }
        }
        lds_fence_wave();
    };

    // ---- forward-mode tangent pass for Jacobian column i (reuses WMMA path) ----
    auto tangent_eval = [&](int i) {
        const float* w0row = sW0 + i * HID;  // W0[i][k]
        v16h a0, a1;
        {
            int lo = (lane < 16) ? 0 : 8;
#pragma unroll
            for (int j = 0; j < 8; ++j) {
                int kA = lo + j, kB = 16 + lo + j;
                a0[j]     = (_Float16)((float)sS0[wv][m * HID + kA]      * w0row[kA]);
                a0[8 + j] = (_Float16)((float)sS0[wv][m * HID + kB]      * w0row[kB]);
                a1[j]     = (_Float16)((float)sS0[wv][m * HID + 32 + kA] * w0row[32 + kA]);
                a1[8 + j] = (_Float16)((float)sS0[wv][m * HID + 32 + kB] * w0row[32 + kB]);
            }
        }
#pragma unroll
        for (int nt = 0; nt < 4; ++nt) {
            v8f acc = {};
            acc = wmma16(a0, frag_ld_b(sW1T, nl + 16 * nt, 0, lane), acc);
            acc = wmma16(a1, frag_ld_b(sW1T, nl + 16 * nt, 32, lane), acc);
            int n = nl + 16 * nt;
#pragma unroll
            for (int v = 0; v < 8; ++v) {
                int mm = v + mh;
                float dh = (float)sS1[wv][mm * HID + n] * acc[v];
                sH[wv][mm * HID + n] = (_Float16)dh;
            }
        }
        lds_fence_wave();
        v16h c0 = frag_ld_a(sH[wv], m, 0, lane);
        v16h c1 = frag_ld_a(sH[wv], m, 32, lane);
#pragma unroll
        for (int nt = 0; nt < 2; ++nt) {
            v8f acc = {};
            acc = wmma16(c0, frag_ld_b(sW2T, nl + 16 * nt, 0, lane), acc);
            acc = wmma16(c1, frag_ld_b(sW2T, nl + 16 * nt, 32, lane), acc);
            int n = nl + 16 * nt;
            if (n < NBONES) {
#pragma unroll
                for (int v = 0; v < 8; ++v)
                    sG[wv][(v + mh) * NBONES + n] = acc[v];
            }
        }
        lds_fence_wave();
    };

    // ---- per-candidate softmax + transform blend (runs in every lane; pair-mirrored) ----
    auto blend = [&](bool save_w, float& f0, float& f1, float& f2) {
        float l[NBONES];
#pragma unroll
        for (int t = 0; t < NBONES / 4; ++t) {
            v4f c4 = *(const v4f*)(&sG[wv][m * NBONES + 4 * t]);  // b128 logits
            l[4 * t + 0] = c4.x; l[4 * t + 1] = c4.y; l[4 * t + 2] = c4.z; l[4 * t + 3] = c4.w;
        }
        float mx = l[0];
#pragma unroll
        for (int n = 1; n < NBONES; ++n) mx = fmaxf(mx, l[n]);
        float wloc[NBONES]; float s = 0.f;
#pragma unroll
        for (int n = 0; n < NBONES; ++n) { wloc[n] = fast_exp2(SB_LOG2E * (l[n] - mx)); s += wloc[n]; }
        float inv = fast_rcp(s);
        v4f bx0 = {0.f, 0.f, 0.f, 0.f}, bx1 = bx0, bx2 = bx0;
#pragma unroll
        for (int n = 0; n < NBONES; ++n) {
            float wn = wloc[n] * inv;
            if (save_w) wgt[n] = wn;
            const v4f* Tn = (const v4f*)(sTf + n * 12);   // 3 x b128 per bone
            bx0 += wn * Tn[0];
            bx1 += wn * Tn[1];
            bx2 += wn * Tn[2];
        }
        f0 = bx0.x * X0 + bx0.y * X1 + bx0.z * X2 + bx0.w;
        f1 = bx1.x * X0 + bx1.y * X1 + bx1.z * X2 + bx1.w;
        f2 = bx2.x * X0 + bx2.y * X1 + bx2.z * X2 + bx2.w;
        if (save_w) {
            Rb[0] = bx0.x; Rb[1] = bx0.y; Rb[2] = bx0.z;
            Rb[3] = bx1.x; Rb[4] = bx1.y; Rb[5] = bx1.z;
            Rb[6] = bx2.x; Rb[7] = bx2.y; Rb[8] = bx2.z;
        }
    };

    // Jacobian column i from tangent logits in sG:
    // Jcol = Rb[:,i] + 20*( sum_n w_n dl_n p_n - (sum_n w_n dl_n) f )
    auto jcol = [&](int i, float f0, float f1, float f2, float* Jc) {
        float dl[NBONES];
#pragma unroll
        for (int t = 0; t < NBONES / 4; ++t) {
            v4f c4 = *(const v4f*)(&sG[wv][m * NBONES + 4 * t]);
            dl[4 * t + 0] = c4.x; dl[4 * t + 1] = c4.y; dl[4 * t + 2] = c4.z; dl[4 * t + 3] = c4.w;
        }
        float sa = 0.f, v0 = 0.f, v1 = 0.f, v2 = 0.f;
#pragma unroll
        for (int n = 0; n < NBONES; ++n) {
            float wd = wgt[n] * dl[n];
            sa += wd;
            const v4f* Tn = (const v4f*)(sTf + n * 12);
            v4f r0 = Tn[0], r1 = Tn[1], r2 = Tn[2];
            float p0 = r0.x * X0 + r0.y * X1 + r0.z * X2 + r0.w;
            float p1 = r1.x * X0 + r1.y * X1 + r1.z * X2 + r1.w;
            float p2 = r2.x * X0 + r2.y * X1 + r2.z * X2 + r2.w;
            v0 += wd * p0; v1 += wd * p1; v2 += wd * p2;
        }
        Jc[0] = Rb[i]     + SOFT_BLEND * (v0 - sa * f0);
        Jc[1] = Rb[3 + i] + SOFT_BLEND * (v1 - sa * f1);
        Jc[2] = Rb[6 + i] + SOFT_BLEND * (v2 - sa * f2);
    };

    // ================= init: g(x_init), analytic Jacobian, J_inv =================
    if (lane < 16) {
        v4f x4 = {X0, X1, X2, 1.f};
        *(v4f*)(&sX[wv][m * 4]) = x4;
    }
    lds_fence_wave();
    mlp_eval(true);

    float F0, F1, F2;
    blend(true, F0, F1, F2);
    float G0 = F0 - T0, G1 = F1 - T1, G2 = F2 - T2;
    // track SQUARED norms: thresholds are positive so compares are equivalent
    float gnopt2 = G0 * G0 + G1 * G1 + G2 * G2;
    bool valid = true;

    float J[9];
    {
        float Jc[3];
        tangent_eval(0); jcol(0, F0, F1, F2, Jc); J[0] = Jc[0]; J[3] = Jc[1]; J[6] = Jc[2];
        tangent_eval(1); jcol(1, F0, F1, F2, Jc); J[1] = Jc[0]; J[4] = Jc[1]; J[7] = Jc[2];
        tangent_eval(2); jcol(2, F0, F1, F2, Jc); J[2] = Jc[0]; J[5] = Jc[1]; J[8] = Jc[2];
    }
    float Ji[9];
    {
        float det = J[0] * (J[4] * J[8] - J[5] * J[7])
                  - J[1] * (J[3] * J[8] - J[5] * J[6])
                  + J[2] * (J[3] * J[7] - J[4] * J[6]);
        float id = fast_rcp(det);
        Ji[0] = (J[4] * J[8] - J[5] * J[7]) * id;
        Ji[1] = (J[2] * J[7] - J[1] * J[8]) * id;
        Ji[2] = (J[1] * J[5] - J[2] * J[4]) * id;
        Ji[3] = (J[5] * J[6] - J[3] * J[8]) * id;
        Ji[4] = (J[0] * J[8] - J[2] * J[6]) * id;
        Ji[5] = (J[2] * J[3] - J[0] * J[5]) * id;
        Ji[6] = (J[3] * J[7] - J[4] * J[6]) * id;
        Ji[7] = (J[1] * J[6] - J[0] * J[7]) * id;
        Ji[8] = (J[0] * J[4] - J[1] * J[3]) * id;
    }
    float U0 = -(Ji[0] * G0 + Ji[1] * G1 + Ji[2] * G2);
    float U1 = -(Ji[3] * G0 + Ji[4] * G1 + Ji[5] * G2);
    float U2 = -(Ji[6] * G0 + Ji[7] * G1 + Ji[8] * G2);
    float DX0 = 0.f, DX1 = 0.f, DX2 = 0.f, DG0 = 0.f, DG1 = 0.f, DG2 = 0.f;

    const float CVG2 = CVG_THRESH * CVG_THRESH;
    const float DVG2 = DVG_THRESH * DVG_THRESH;

    // ================= Broyden loop (fixed 20 steps, matching lax.scan) =================
#pragma unroll 1
    for (int it = 0; it < MAX_STEPS; ++it) {
        bool mv = valid;
        if (mv) { DX0 = U0; DX1 = U1; DX2 = U2; X0 += DX0; X1 += DX1; X2 += DX2; }
        if (lane < 16) {
            v4f x4 = {X0, X1, X2, 1.f};
            *(v4f*)(&sX[wv][m * 4]) = x4;
        }
        lds_fence_wave();
        mlp_eval(false);
        float f0, f1, f2;
        blend(false, f0, f1, f2);
        float g0 = f0 - T0, g1 = f1 - T1, g2 = f2 - T2;
        if (mv) { DG0 = g0 - G0; DG1 = g1 - G1; DG2 = g2 - G2; G0 = g0; G1 = g1; G2 = g2; }
        float gn2 = G0 * G0 + G1 * G1 + G2 * G2;
        gnopt2 = fminf(gnopt2, gn2);
        valid = (gnopt2 > CVG2) && (gn2 < DVG2);
        if (valid) {
            float vT0 = DX0 * Ji[0] + DX1 * Ji[3] + DX2 * Ji[6];
            float vT1 = DX0 * Ji[1] + DX1 * Ji[4] + DX2 * Ji[7];
            float vT2 = DX0 * Ji[2] + DX1 * Ji[5] + DX2 * Ji[8];
            float a0v = DX0 - (Ji[0] * DG0 + Ji[1] * DG1 + Ji[2] * DG2);
            float a1v = DX1 - (Ji[3] * DG0 + Ji[4] * DG1 + Ji[5] * DG2);
            float a2v = DX2 - (Ji[6] * DG0 + Ji[7] * DG1 + Ji[8] * DG2);
            float bden = vT0 * DG0 + vT1 * DG1 + vT2 * DG2;
            bden += (bden >= 0.f) ? BR_EPS : -BR_EPS;
            float ib = fast_rcp(bden);
            Ji[0] += a0v * vT0 * ib; Ji[1] += a0v * vT1 * ib; Ji[2] += a0v * vT2 * ib;
            Ji[3] += a1v * vT0 * ib; Ji[4] += a1v * vT1 * ib; Ji[5] += a1v * vT2 * ib;
            Ji[6] += a2v * vT0 * ib; Ji[7] += a2v * vT1 * ib; Ji[8] += a2v * vT2 * ib;
        }
        U0 = -(Ji[0] * G0 + Ji[1] * G1 + Ji[2] * G2);
        U1 = -(Ji[3] * G0 + Ji[4] * G1 + Ji[5] * G2);
        U2 = -(Ji[6] * G0 + Ji[7] * G1 + Ji[8] * G2);
    }

    // output = xc_opt (correction term is exactly zero in the forward value)
    if (lane < 16 && cand < P) {
        out[cand * 3 + 0] = X0;
        out[cand * 3 + 1] = X1;
        out[cand * 3 + 2] = X2;
    }
}

extern "C" void kernel_launch(void* const* d_in, const int* in_sizes, int n_in,
                              void* d_out, int out_size, void* d_ws, size_t ws_size,
                              hipStream_t stream) {
    (void)n_in; (void)d_ws; (void)ws_size; (void)out_size;
    const float* xd  = (const float*)d_in[0];
    const float* tfs = (const float*)d_in[1];
    const float* W0  = (const float*)d_in[2];
    const float* b0  = (const float*)d_in[3];
    const float* W1  = (const float*)d_in[4];
    const float* b1  = (const float*)d_in[5];
    const float* W2  = (const float*)d_in[6];
    const float* b2  = (const float*)d_in[7];
    float* out = (float*)d_out;

    const int Npts = in_sizes[0] / 3;      // B*N points
    const int P    = Npts * NINIT;         // candidates
    const int blocks = (P + CAND_PER_BLOCK - 1) / CAND_PER_BLOCK;

    hipLaunchKernelGGL(fd_broyden_kernel, dim3(blocks), dim3(NW * 32), 0, stream,
                       xd, tfs, W0, b0, W1, b1, W2, b2, out, P);
}
